// LSTM3lr_Decoder_76690936037399
// MI455X (gfx1250) — matrix operand order repaired and checked by reading, not compiled, not run on hardware
//
#include <hip/hip_runtime.h>

typedef __attribute__((ext_vector_type(16))) __bf16 v16bf;
typedef __attribute__((ext_vector_type(8)))  __bf16 v8bf;
typedef __attribute__((ext_vector_type(8)))  float  v8f;

static __device__ __forceinline__ __bf16 f2bf(float f) {
  unsigned u = __builtin_bit_cast(unsigned, f);
  unsigned r = u + 0x7FFFu + ((u >> 16) & 1u);   // round-to-nearest-even
  unsigned short h = (unsigned short)(r >> 16);
  return __builtin_bit_cast(__bf16, h);
}

static __device__ __forceinline__ float sigmoidf_(float x) {
  return 1.0f / (1.0f + __expf(-x));
}

// Force a wave-uniform value back into SGPRs so loop bounds / base addresses
// use scalar control flow and saddr-form loads.
static __device__ __forceinline__ int uniform_i(int v) {
  return __builtin_amdgcn_readfirstlane(v);
}
static __device__ __forceinline__ const __bf16* uniform_p(const __bf16* p) {
  unsigned long long u = (unsigned long long)p;
  unsigned lo = __builtin_amdgcn_readfirstlane((unsigned)u);
  unsigned hi = __builtin_amdgcn_readfirstlane((unsigned)(u >> 32));
  return (const __bf16*)(((unsigned long long)hi << 32) | lo);
}

// A-fragment (16x32 bf16, M x K), contiguous k order matching B:
// lane<16: m=lane,  K = 0..15
// lane>=16: m=lane-16, K = 16..31
static __device__ __forceinline__ v16bf load_fragA(const __bf16* __restrict__ base,
                                                   int ld, int m0, int k0, int lane) {
  int m  = m0 + (lane & 15);
  int kb = k0 + ((lane < 16) ? 0 : 16);
  const __bf16* p = base + (size_t)m * ld + kb;
  v8bf lo = *(const v8bf*)(p);        // k = kb .. kb+7
  v8bf hi = *(const v8bf*)(p + 8);    // k = kb+8 .. kb+15
  return __builtin_shufflevector(lo, hi, 0,1,2,3,4,5,6,7,8,9,10,11,12,13,14,15);
}

// B-fragment (32x16 bf16, K x N): lane<16: n=lane, K=0..15; lane>=16: n=lane-16, K=16..31
static __device__ __forceinline__ v16bf load_fragB(const __bf16* __restrict__ Wg,
                                                   int ld, int n0, int k0, int lane) {
  int n  = n0 + (lane & 15);
  int kb = k0 + ((lane < 16) ? 0 : 16);
  return *(const v16bf*)(Wg + (size_t)n * ld + kb);
}

// ---------------------------------------------------------------------------
// Fused LSTM cell: gates = A1 @ W1^T + A2 @ W2^T + bias  (per-gate padded bf16
// weights [4][1024][ka]); elementwise LSTM update fused in-register.
// Grid: 64 blocks x 256 threads. Block = (batch group of 32) x (64 hidden cols).
// Waves 0-3 do GEMM1, waves 4-7 do GEMM2 for the same output tiles; LDS reduce.
// ---------------------------------------------------------------------------
__global__ __launch_bounds__(256) void lstm_cell_kernel(
    const __bf16* __restrict__ A1, const __bf16* __restrict__ W1, int ka1,
    const __bf16* __restrict__ A2, const __bf16* __restrict__ W2,
    const float*  __restrict__ bias,     // [4][1024]
    float*        __restrict__ c_state,  // [128][1024]
    float*        __restrict__ h_out,    // d_out h slice; row stride 25000
    __bf16*       __restrict__ h_next,   // [128][1024] bf16 (ping-pong)
    __bf16*       __restrict__ h_hist)   // optional layer-2 history, or null
{
  __shared__ float red[4][32][64];
  const int lane    = threadIdx.x & 31;
  const int wave    = threadIdx.x >> 5;
  const int pairIdx = wave & 3;
  const bool second = wave >= 4;
  const int mg    = blockIdx.x >> 4;          // 0..3  batch group
  const int kcg   = blockIdx.x & 15;          // 0..15 column group
  const int n0    = (kcg * 4 + pairIdx) * 16; // hidden col tile base (0..1008)
  const int mbase = mg * 32;

  const v8f vzero = {0.f,0.f,0.f,0.f,0.f,0.f,0.f,0.f};
  v8f acc[2][4];
  #pragma unroll
  for (int mt = 0; mt < 2; ++mt)
    #pragma unroll
    for (int g = 0; g < 4; ++g) acc[mt][g] = vzero;

  // Wave-uniform GEMM operands, forced scalar for clean codegen.
  const __bf16* A  = uniform_p(second ? A2 : A1);
  const __bf16* W  = uniform_p(second ? W2 : W1);
  const int     ka = uniform_i(second ? 1024 : ka1);
  const size_t  gstep = (size_t)1024 * ka;    // per-gate stride in elements

  for (int k0 = 0; k0 < ka; k0 += 32) {
    v16bf a0 = load_fragA(A, ka, mbase,      k0, lane);
    v16bf a1 = load_fragA(A, ka, mbase + 16, k0, lane);
    v16bf b0 = load_fragB(W,             ka, n0, k0, lane);
    v16bf b1 = load_fragB(W +     gstep, ka, n0, k0, lane);
    v16bf b2 = load_fragB(W + 2 * gstep, ka, n0, k0, lane);
    v16bf b3 = load_fragB(W + 3 * gstep, ka, n0, k0, lane);
    if (k0 + 32 < ka)
      __builtin_prefetch(W + (size_t)n0 * ka + k0 + 32, 0, 1);
    acc[0][0] = __builtin_amdgcn_wmma_f32_16x16x32_bf16(false, a0, false, b0, (short)0, acc[0][0], false, false);
    acc[1][0] = __builtin_amdgcn_wmma_f32_16x16x32_bf16(false, a1, false, b0, (short)0, acc[1][0], false, false);
    acc[0][1] = __builtin_amdgcn_wmma_f32_16x16x32_bf16(false, a0, false, b1, (short)0, acc[0][1], false, false);
    acc[1][1] = __builtin_amdgcn_wmma_f32_16x16x32_bf16(false, a1, false, b1, (short)0, acc[1][1], false, false);
    acc[0][2] = __builtin_amdgcn_wmma_f32_16x16x32_bf16(false, a0, false, b2, (short)0, acc[0][2], false, false);
    acc[1][2] = __builtin_amdgcn_wmma_f32_16x16x32_bf16(false, a1, false, b2, (short)0, acc[1][2], false, false);
    acc[0][3] = __builtin_amdgcn_wmma_f32_16x16x32_bf16(false, a0, false, b3, (short)0, acc[0][3], false, false);
    acc[1][3] = __builtin_amdgcn_wmma_f32_16x16x32_bf16(false, a1, false, b3, (short)0, acc[1][3], false, false);
  }

  if (second) {
    #pragma unroll
    for (int mt = 0; mt < 2; ++mt)
      #pragma unroll
      for (int g = 0; g < 4; ++g)
        #pragma unroll
        for (int v = 0; v < 8; ++v)
          red[pairIdx][lane][(mt*4+g)*8+v] = acc[mt][g][v];
  }
  __syncthreads();

  if (!second) {
    const int  n     = n0 + (lane & 15);
    const bool valid = (n < 1000);
    const float bi = bias[n],        bf_ = bias[1024+n];
    const float bg = bias[2048+n],   bo  = bias[3072+n];
    #pragma unroll
    for (int mt = 0; mt < 2; ++mt) {
      #pragma unroll
      for (int v = 0; v < 8; ++v) {
        const int m = mbase + mt*16 + ((lane < 16) ? v : v + 8);
        float gi = acc[mt][0][v] + red[pairIdx][lane][(mt*4+0)*8+v] + bi;
        float gf = acc[mt][1][v] + red[pairIdx][lane][(mt*4+1)*8+v] + bf_;
        float gg = acc[mt][2][v] + red[pairIdx][lane][(mt*4+2)*8+v] + bg;
        float go = acc[mt][3][v] + red[pairIdx][lane][(mt*4+3)*8+v] + bo;
        float c_old = c_state[(size_t)m*1024 + n];
        float cn = sigmoidf_(gf) * c_old + sigmoidf_(gi) * tanhf(gg);
        float hn = sigmoidf_(go) * tanhf(cn);
        c_state[(size_t)m*1024 + n] = cn;
        __bf16 hb = valid ? f2bf(hn) : f2bf(0.0f);
        h_next[(size_t)m*1024 + n] = hb;
        if (h_hist) h_hist[(size_t)m*1024 + n] = hb;
        if (valid)  h_out[(size_t)m*25000 + n] = hn;
      }
    }
  }
}

// Final batched projection: out[b,t,:] = h2[t,b,:] @ W_out + b_out
// M = 25*128 = 3200 rows, N = 64 (60 padded), K = 1024. Grid 25x256.
__global__ __launch_bounds__(256) void proj_kernel(
    const __bf16* __restrict__ h2,   // [25*128][1024]
    const __bf16* __restrict__ Wo,   // [64][1024]  (Wo[j][k] = W_out[k][j])
    const float*  __restrict__ bout, // [60]
    float*        __restrict__ out)  // [128][25][60]
{
  const int lane = threadIdx.x & 31;
  const int wave = threadIdx.x >> 5;
  const int mt   = blockIdx.x * 8 + wave;   // 0..199
  const int r0   = mt * 16;
  const v8f vzero = {0.f,0.f,0.f,0.f,0.f,0.f,0.f,0.f};
  v8f acc[4] = {vzero, vzero, vzero, vzero};
  for (int k0 = 0; k0 < 1024; k0 += 32) {
    v16bf a  = load_fragA(h2, 1024, r0, k0, lane);
    v16bf b0 = load_fragB(Wo, 1024,  0, k0, lane);
    v16bf b1 = load_fragB(Wo, 1024, 16, k0, lane);
    v16bf b2 = load_fragB(Wo, 1024, 32, k0, lane);
    v16bf b3 = load_fragB(Wo, 1024, 48, k0, lane);
    acc[0] = __builtin_amdgcn_wmma_f32_16x16x32_bf16(false, a, false, b0, (short)0, acc[0], false, false);
    acc[1] = __builtin_amdgcn_wmma_f32_16x16x32_bf16(false, a, false, b1, (short)0, acc[1], false, false);
    acc[2] = __builtin_amdgcn_wmma_f32_16x16x32_bf16(false, a, false, b2, (short)0, acc[2], false, false);
    acc[3] = __builtin_amdgcn_wmma_f32_16x16x32_bf16(false, a, false, b3, (short)0, acc[3], false, false);
  }
  #pragma unroll
  for (int nt = 0; nt < 4; ++nt) {
    int j = nt*16 + (lane & 15);
    if (j < 60) {
      #pragma unroll
      for (int v = 0; v < 8; ++v) {
        int r  = r0 + ((lane < 16) ? v : v + 8);
        int b  = r & 127;
        int ts = r >> 7;
        out[((size_t)b*25 + ts)*60 + j] = acc[nt][v] + bout[j];
      }
    }
  }
}

// ------------------------- preparation kernels -----------------------------

// fp32 (4000 x K) -> bf16 per-gate padded [4][1024][Kp]
__global__ void pack_w_kernel(const float* __restrict__ W, __bf16* __restrict__ out,
                              int K, int Kp) {
  int idx = blockIdx.x * blockDim.x + threadIdx.x;
  int k = idx % Kp;
  int n = (idx / Kp) & 1023;
  int g = idx / (Kp * 1024);
  if (g >= 4) return;
  float v = (n < 1000 && k < K) ? W[(size_t)(g*1000 + n)*K + k] : 0.0f;
  out[idx] = f2bf(v);
}

// Wc[g][n][k] = sum_j W_ih0[g*1000+n][j] * W_out[k][j]  (folds W_out into layer0)
__global__ void compute_wc_kernel(const float* __restrict__ Wih0,
                                  const float* __restrict__ Wout,
                                  __bf16* __restrict__ out) {
  int idx = blockIdx.x * blockDim.x + threadIdx.x;
  int k = idx & 1023;
  int n = (idx >> 10) & 1023;
  int g = idx >> 20;
  if (g >= 4) return;
  float s = 0.0f;
  if (n < 1000 && k < 1000) {
    const float* wi = Wih0 + (size_t)(g*1000 + n)*60;
    const float* wo = Wout + (size_t)k*60;
    #pragma unroll 4
    for (int j = 0; j < 60; ++j) s += wi[j] * wo[j];
  }
  out[idx] = f2bf(s);
}

// Wo[j][k] = W_out[k][j], padded to [64][1024]
__global__ void pack_wo_kernel(const float* __restrict__ Wout, __bf16* __restrict__ out) {
  int idx = blockIdx.x * blockDim.x + threadIdx.x;
  int k = idx & 1023;
  int j = idx >> 10;
  if (j >= 64) return;
  float v = (j < 60 && k < 1000) ? Wout[(size_t)k*60 + j] : 0.0f;
  out[idx] = f2bf(v);
}

// x0[b][j] = dec_in[b][0][j], padded to [128][64]
__global__ void pack_x0_kernel(const float* __restrict__ dec, __bf16* __restrict__ out) {
  int idx = blockIdx.x * blockDim.x + threadIdx.x;
  if (idx >= 128*64) return;
  int j = idx & 63, b = idx >> 6;
  out[idx] = f2bf(j < 60 ? dec[(size_t)b*60 + j] : 0.0f);
}

// bias[g][n] = b1 + b2 (+ sum_j bout[j]*Wih0[g*1000+n][j] when folding W_out)
__global__ void pack_bias_kernel(const float* __restrict__ b1, const float* __restrict__ b2,
                                 const float* __restrict__ Wih0, const float* __restrict__ bout,
                                 float* __restrict__ out) {
  int idx = blockIdx.x * blockDim.x + threadIdx.x;
  if (idx >= 4096) return;
  int n = idx & 1023, g = idx >> 10;
  float v = 0.0f;
  if (n < 1000) {
    int s = g*1000 + n;
    v = b1[s] + b2[s];
    if (Wih0) {
      const float* w = Wih0 + (size_t)s*60;
      for (int j = 0; j < 60; ++j) v += bout[j] * w[j];
    }
  }
  out[idx] = v;
}

__global__ void init_state_kernel(__bf16* __restrict__ hb, float* __restrict__ c) {
  int idx = blockIdx.x * blockDim.x + threadIdx.x;
  if (idx < 6*128*1024) hb[idx] = f2bf(0.0f);
  if (idx < 3*128*1024) c[idx] = 0.0f;
}

// ---------------------------------------------------------------------------

extern "C" void kernel_launch(void* const* d_in, const int* in_sizes, int n_in,
                              void* d_out, int out_size, void* d_ws, size_t ws_size,
                              hipStream_t stream) {
  (void)in_sizes; (void)n_in; (void)out_size; (void)ws_size;
  const float* dec_in = (const float*)d_in[1];   // enc_state (d_in[0]) is unused by reference
  const float* W_ih0  = (const float*)d_in[2];
  const float* W_hh0  = (const float*)d_in[3];
  const float* b_ih0  = (const float*)d_in[4];
  const float* b_hh0  = (const float*)d_in[5];
  const float* W_ih1  = (const float*)d_in[6];
  const float* W_hh1  = (const float*)d_in[7];
  const float* b_ih1  = (const float*)d_in[8];
  const float* b_hh1  = (const float*)d_in[9];
  const float* W_ih2  = (const float*)d_in[10];
  const float* W_hh2  = (const float*)d_in[11];
  const float* b_ih2  = (const float*)d_in[12];
  const float* b_hh2  = (const float*)d_in[13];
  const float* W_out  = (const float*)d_in[14];
  const float* b_out  = (const float*)d_in[15];

  float* out      = (float*)d_out;
  float* h_region = out + (size_t)128*25*60;

  char* p = (char*)d_ws;
  auto alloc = [&](size_t bytes) { char* r = p; p += (bytes + 255) & ~(size_t)255; return r; };
  const size_t WBE = (size_t)4*1024*1024;          // elements per padded big weight
  __bf16* Wc    = (__bf16*)alloc(WBE*2);
  __bf16* Whh0  = (__bf16*)alloc(WBE*2);
  __bf16* Wih1  = (__bf16*)alloc(WBE*2);
  __bf16* Whh1  = (__bf16*)alloc(WBE*2);
  __bf16* Wih2  = (__bf16*)alloc(WBE*2);
  __bf16* Whh2  = (__bf16*)alloc(WBE*2);
  __bf16* Wih0p = (__bf16*)alloc((size_t)4*1024*64*2);
  __bf16* Wo    = (__bf16*)alloc((size_t)64*1024*2);
  __bf16* x0    = (__bf16*)alloc((size_t)128*64*2);
  __bf16* h_b   = (__bf16*)alloc((size_t)6*128*1024*2);   // 2 parities x 3 layers
  __bf16* h2h   = (__bf16*)alloc((size_t)25*128*1024*2);  // layer-2 history
  float*  c_st  = (float*)alloc((size_t)3*128*1024*4);
  float*  bias00= (float*)alloc(4096*4);
  float*  bias0 = (float*)alloc(4096*4);
  float*  bias1 = (float*)alloc(4096*4);
  float*  bias2 = (float*)alloc(4096*4);

  const int BW = (4*1024*1024)/256;
  pack_w_kernel<<<BW, 256, 0, stream>>>(W_hh0, Whh0, 1000, 1024);
  pack_w_kernel<<<BW, 256, 0, stream>>>(W_ih1, Wih1, 1000, 1024);
  pack_w_kernel<<<BW, 256, 0, stream>>>(W_hh1, Whh1, 1000, 1024);
  pack_w_kernel<<<BW, 256, 0, stream>>>(W_ih2, Wih2, 1000, 1024);
  pack_w_kernel<<<BW, 256, 0, stream>>>(W_hh2, Whh2, 1000, 1024);
  pack_w_kernel<<<(4*1024*64)/256, 256, 0, stream>>>(W_ih0, Wih0p, 60, 64);
  compute_wc_kernel<<<BW, 256, 0, stream>>>(W_ih0, W_out, Wc);
  pack_wo_kernel<<<(64*1024)/256, 256, 0, stream>>>(W_out, Wo);
  pack_x0_kernel<<<(128*64)/256, 256, 0, stream>>>(dec_in, x0);
  pack_bias_kernel<<<16, 256, 0, stream>>>(b_ih0, b_hh0, nullptr, nullptr, bias00);
  pack_bias_kernel<<<16, 256, 0, stream>>>(b_ih0, b_hh0, W_ih0, b_out, bias0);
  pack_bias_kernel<<<16, 256, 0, stream>>>(b_ih1, b_hh1, nullptr, nullptr, bias1);
  pack_bias_kernel<<<16, 256, 0, stream>>>(b_ih2, b_hh2, nullptr, nullptr, bias2);
  init_state_kernel<<<(6*128*1024)/256, 256, 0, stream>>>(h_b, c_st);

  const size_t HS = (size_t)128*1024;
  auto hb = [&](int L, int par) { return h_b + ((size_t)par*3 + L)*HS; };

  for (int t = 0; t < 25; ++t) {
    const int pw = t & 1, pr = pw ^ 1;
    float* hout0 = h_region + ((size_t)0*128*25 + t)*1000;
    float* hout1 = h_region + ((size_t)1*128*25 + t)*1000;
    float* hout2 = h_region + ((size_t)2*128*25 + t)*1000;
    if (t == 0)
      lstm_cell_kernel<<<64, 256, 0, stream>>>(x0, Wih0p, 64,
          hb(0,pr), Whh0, bias00, c_st + 0*HS, hout0, hb(0,pw), (__bf16*)nullptr);
    else
      lstm_cell_kernel<<<64, 256, 0, stream>>>(hb(2,pr), Wc, 1024,
          hb(0,pr), Whh0, bias0, c_st + 0*HS, hout0, hb(0,pw), (__bf16*)nullptr);
    lstm_cell_kernel<<<64, 256, 0, stream>>>(hb(0,pw), Wih1, 1024,
        hb(1,pr), Whh1, bias1, c_st + 1*HS, hout1, hb(1,pw), (__bf16*)nullptr);
    lstm_cell_kernel<<<64, 256, 0, stream>>>(hb(1,pw), Wih2, 1024,
        hb(2,pr), Whh2, bias2, c_st + 2*HS, hout2, hb(2,pw), h2h + (size_t)t*HS);
  }
  proj_kernel<<<25, 256, 0, stream>>>(h2h, Wo, b_out, out);
}
